// OnlineHardTripletLoss_85504208929326
// MI455X (gfx1250) — compile-verified, matrix operand order checked by reading, not run on hardware
//
#include <hip/hip_runtime.h>
#include <math.h>

// Problem constants from the reference
#define BSZ   384
#define DDIM  256
#define MARGIN_F 0.3f
#define HARD_R   0.5f
#define SEMI_R   0.3f

typedef __attribute__((ext_vector_type(2)))  float    v2f;
typedef __attribute__((ext_vector_type(8)))  float    v8f;
typedef __attribute__((ext_vector_type(16))) _Float16 v16h;

#if defined(__has_builtin)
#if __has_builtin(__builtin_amdgcn_wmma_f32_16x16x4_f32)
#define HAVE_WMMA_F32_K4 1
#endif
#endif

// ---------------------------------------------------------------------------
// Zero the global accumulators (total, count) every launch (graph-replayable).
__global__ void init_acc_kernel(float* __restrict__ acc) {
    if (threadIdx.x == 0) { acc[0] = 0.0f; acc[1] = 0.0f; }
}

// ---------------------------------------------------------------------------
// sq[r] = ||E[r,:]||^2   (one thread per row; 384 rows, trivially cheap)
__global__ void row_sq_kernel(const float* __restrict__ E, float* __restrict__ sq) {
    int r = blockIdx.x * blockDim.x + threadIdx.x;
    if (r >= BSZ) return;
    const float* p = E + (size_t)r * DDIM;
    float s = 0.0f;
    for (int k = 0; k < DDIM; ++k) { float v = p[k]; s = fmaf(v, v, s); }
    sq[r] = s;
}

// ---------------------------------------------------------------------------
// dist[i][j] = sqrt(relu(sq[i] + sq[j] - 2*(E E^T)[i][j]) + 1e-16)
// One wave per 16x16 tile. Gram matrix via f32 WMMA (16x16x4), K=256 -> 64 steps.
__global__ void __launch_bounds__(32)
dist_wmma_kernel(const float* __restrict__ E, const float* __restrict__ sq,
                 float* __restrict__ dist) {
    const int tm   = blockIdx.x;        // tile row (M block)
    const int tn   = blockIdx.y;        // tile col (N block)
    const int lane = threadIdx.x;       // 0..31 (wave32)
    const int hf   = lane >> 4;         // half-wave: 0 or 1
    const int l16  = lane & 15;

    const int rowA = tm * 16 + l16;     // A: rows of E (M)
    const int rowB = tn * 16 + l16;     // B col n of E^T == row n of E (N)

    v8f acc = {};

#if HAVE_WMMA_F32_K4
    // f32 16x16x4 fragment layout:
    //  A: lanes 0-15 hold M=l16, K = kb+0 (v0), kb+1 (v1);
    //     lanes 16-31 hold M=l16, K = kb+2 (v0), kb+3 (v1).
    //  B mirrors A with N in place of M -> identical per-lane float2 loads.
    const int koff = hf * 2;
    const float* pa = E + (size_t)rowA * DDIM + koff;
    const float* pb = E + (size_t)rowB * DDIM + koff;
    for (int kb = 0; kb < DDIM; kb += 4) {
        v2f a, b;
        a.x = pa[kb];  a.y = pa[kb + 1];
        b.x = pb[kb];  b.y = pb[kb + 1];
        acc = __builtin_amdgcn_wmma_f32_16x16x4_f32(
            false, a, false, b, (short)0, acc, false, false);
    }
#else
    // Fallback: f16 16x16x32 WMMA (precision-compromised but codegen-confirmed).
    // 16-bit A layout: VGPR v in 0..3 -> K = hf*8 + 2v,2v+1 ; v in 4..7 -> K = 16 + hf*8 + ...
    for (int kb = 0; kb < DDIM; kb += 32) {
        v16h a, b;
#pragma unroll
        for (int e = 0; e < 16; ++e) {
            const int K = (e < 8) ? (hf * 8 + e) : (16 + hf * 8 + (e - 8));
            a[e] = (_Float16)E[(size_t)rowA * DDIM + kb + K];
            b[e] = (_Float16)E[(size_t)rowB * DDIM + kb + K];
        }
        acc = __builtin_amdgcn_wmma_f32_16x16x32_f16(
            false, a, false, b, (short)0, acc, false, false);
    }
#endif

    // C/D layout: VGPR r holds M = r + 8*hf, N = l16.
#pragma unroll
    for (int r = 0; r < 8; ++r) {
        const int gi = tm * 16 + r + 8 * hf;
        const int gj = tn * 16 + l16;
        float d2 = sq[gi] + sq[gj] - 2.0f * acc[r];
        d2 = d2 > 0.0f ? d2 : 0.0f;
        dist[(size_t)gi * BSZ + gj] = sqrtf(d2 + 1e-16f);
    }
}

// ---------------------------------------------------------------------------
// Triplet selection. One block (8 waves) per anchor i. Each wave owns a stripe
// of positives j; the in-order cumsum-limited negative selection is done 32 k's
// at a time with ballot + lane-prefix popcount (exclusive prefix < limit).
__global__ void __launch_bounds__(256)
select_kernel(const float* __restrict__ dist, const int* __restrict__ labels,
              float* __restrict__ acc) {
    const int i    = blockIdx.x;
    const int lane = threadIdx.x & 31;
    const int wv   = threadIdx.x >> 5;   // 0..7
    const int li   = labels[i];

    // neg_count for this anchor (each wave computes it redundantly; cheap)
    int negc = 0;
    for (int k = lane; k < BSZ; k += 32) negc += (labels[k] != li) ? 1 : 0;
#pragma unroll
    for (int off = 16; off > 0; off >>= 1) negc += __shfl_xor(negc, off, 32);

    const int num_hard = max(1, (int)floorf((float)negc * HARD_R));
    const int num_semi = max(1, (int)floorf((float)negc * SEMI_R));

    const float* di = dist + (size_t)i * BSZ;
    float lsum = 0.0f, lcnt = 0.0f;

    for (int j = wv; j < BSZ; j += 8) {
        if (j == i || labels[j] != li) continue;   // pos_mask
        const float ap = di[j];
        int hbase = 0, sbase = 0;
        for (int kb = 0; kb < BSZ; kb += 32) {
            const int  k   = kb + lane;
            const float an = di[k];
            const bool nm   = (labels[k] != li);
            const bool hard = nm && (an < ap);
            const bool semi = nm && (an > ap) && (an < ap + MARGIN_F);

            const unsigned long long hm = __ballot(hard);
            const unsigned long long sm = __ballot(semi);
            const unsigned long long below = (1ull << lane) - 1ull;
            const int ph = __popcll(hm & below);   // exclusive prefix
            const int ps = __popcll(sm & below);

            const bool sel = (hard && (hbase + ph) < num_hard) ||
                             (semi && (sbase + ps) < num_semi);
            if (sel) {
                float t = ap - an + MARGIN_F;
                lsum += t > 0.0f ? t : 0.0f;
                lcnt += 1.0f;
            }
            hbase += __popcll(hm);
            sbase += __popcll(sm);
        }
    }

#pragma unroll
    for (int off = 16; off > 0; off >>= 1) {
        lsum += __shfl_xor(lsum, off, 32);
        lcnt += __shfl_xor(lcnt, off, 32);
    }
    if (lane == 0 && (lcnt != 0.0f)) {
        atomicAdd(&acc[0], lsum);
        atomicAdd(&acc[1], lcnt);
    }
}

// ---------------------------------------------------------------------------
__global__ void finalize_kernel(const float* __restrict__ acc, float* __restrict__ out) {
    if (threadIdx.x == 0 && blockIdx.x == 0)
        out[0] = (acc[1] > 0.0f) ? (acc[0] / acc[1]) : 0.0f;
}

// ---------------------------------------------------------------------------
extern "C" void kernel_launch(void* const* d_in, const int* in_sizes, int n_in,
                              void* d_out, int out_size, void* d_ws, size_t ws_size,
                              hipStream_t stream) {
    const float* E      = (const float*)d_in[0];   // (384,256) f32
    const int*   labels = (const int*)d_in[1];     // (384,) int
    float*       out    = (float*)d_out;           // scalar f32

    // Workspace layout (floats): [0..1] accumulators, [64..] sq (384), [512..] dist (384*384)
    float* ws   = (float*)d_ws;
    float* acc  = ws;
    float* sq   = ws + 64;
    float* dist = ws + 512;

    init_acc_kernel<<<1, 32, 0, stream>>>(acc);
    row_sq_kernel<<<(BSZ + 127) / 128, 128, 0, stream>>>(E, sq);
    dist_wmma_kernel<<<dim3(BSZ / 16, BSZ / 16), 32, 0, stream>>>(E, sq, dist);
    select_kernel<<<BSZ, 256, 0, stream>>>(dist, labels, acc);
    finalize_kernel<<<1, 32, 0, stream>>>(acc, out);
}
